// ProteinFeatures_63187558859337
// MI455X (gfx1250) — compile-verified, matrix operand order checked by reading, not run on hardware
//
#include <hip/hip_runtime.h>
#include <hip/hip_bf16.h>

typedef __attribute__((ext_vector_type(16))) _Float16 v16h;
typedef __attribute__((ext_vector_type(8)))  _Float16 v8h;
typedef __attribute__((ext_vector_type(8)))  float    v8f;

#define TOPK    48
#define FEAT    416          // 16 pos + 25*16 RBF
#define FSTRIDE 424          // padded LDS row stride (multiple of 8 halves)
#define EDIM    128

// pair atom indices (N=0, Ca=1, C=2, O=3, Cb=4), A-of-i / B-of-neighbor
__device__ __constant__ int c_pa[24] = {0,2,3,4,1,1,1,1,0,0,0,4,4,3,0,2,3,4,2,3,4,2,3,2};
__device__ __constant__ int c_pb[24] = {0,2,3,4,0,2,3,4,2,3,4,2,3,2,1,1,1,1,0,0,0,4,4,3};

// ---------------- kernel 1: per-residue atom pack + Cb ----------------
__global__ void k_coords(const float* __restrict__ X, float* __restrict__ coords, int BL) {
    int t = blockIdx.x * blockDim.x + threadIdx.x;
    if (t >= BL) return;
    const float* p = X + (size_t)t * 12;
    float N[3], Ca[3], C[3], O[3];
#pragma unroll
    for (int c = 0; c < 3; ++c) { N[c]=p[c]; Ca[c]=p[3+c]; C[c]=p[6+c]; O[c]=p[9+c]; }
    float b[3], cc[3], a[3];
#pragma unroll
    for (int c = 0; c < 3; ++c) { b[c] = Ca[c]-N[c]; cc[c] = C[c]-Ca[c]; }
    a[0] = b[1]*cc[2] - b[2]*cc[1];
    a[1] = b[2]*cc[0] - b[0]*cc[2];
    a[2] = b[0]*cc[1] - b[1]*cc[0];
    float* o = coords + (size_t)t * 15;
#pragma unroll
    for (int c = 0; c < 3; ++c) {
        o[c]    = N[c];
        o[3+c]  = Ca[c];
        o[6+c]  = C[c];
        o[9+c]  = O[c];
        o[12+c] = -0.58273431f*a[c] + 0.56802827f*b[c] - 0.54067466f*cc[c] + Ca[c];
    }
}

// ---------------- kernel 2: W_edge -> f16 ----------------
__global__ void k_whalf(const float* __restrict__ W, _Float16* __restrict__ Wh, int n) {
    int t = blockIdx.x * blockDim.x + threadIdx.x;
    if (t < n) Wh[t] = (_Float16)W[t];
}

// ---------------- kernel 3: top-K neighbors (exact tie-break like jax top_k) ----------------
__device__ inline void warpMinPair(unsigned& h, unsigned& l) {
#pragma unroll
    for (int off = 16; off > 0; off >>= 1) {
        unsigned oh = __shfl_xor(h, off, 32);
        unsigned ol = __shfl_xor(l, off, 32);
        if (oh < h || (oh == h && ol < l)) { h = oh; l = ol; }
    }
}

__global__ __launch_bounds__(1024) void k_topk(const float* __restrict__ coords,
                                               const float* __restrict__ mask,
                                               float* __restrict__ Dn,
                                               int* __restrict__ Eidx,
                                               int L) {
    int bl = blockIdx.x;
    int b  = bl / L;
    int i  = bl % L;
    int j  = threadIdx.x;          // blockDim.x == L == 1024
    int lane = j & 31, wid = j >> 5;

    __shared__ float    s_cai[3];
    __shared__ float    s_wmax[32];
    __shared__ unsigned s_wh[32], s_wl[32];
    __shared__ unsigned s_bcast;

    if (j < 3) s_cai[j] = coords[(size_t)bl*15 + 3 + j];
    __syncthreads();

    const float* cj = coords + ((size_t)(b * L + j)) * 15 + 3;
    float dx = s_cai[0]-cj[0], dy = s_cai[1]-cj[1], dz = s_cai[2]-cj[2];
    float dist = sqrtf(dx*dx + dy*dy + dz*dz + 1e-6f);
    float m2 = mask[b*L + i] * mask[b*L + j];
    float D  = m2 * dist;

    // row max of D (for D_adjust)
    float mx = D;
#pragma unroll
    for (int off = 16; off > 0; off >>= 1) mx = fmaxf(mx, __shfl_xor(mx, off, 32));
    if (lane == 0) s_wmax[wid] = mx;
    __syncthreads();
    if (wid == 0) {
        float v = s_wmax[lane];
#pragma unroll
        for (int off = 16; off > 0; off >>= 1) v = fmaxf(v, __shfl_xor(v, off, 32));
        if (lane == 0) s_wmax[0] = v;
    }
    __syncthreads();
    float Dadj = D + (1.0f - m2) * s_wmax[0];

    unsigned hb = __float_as_uint(Dadj);   // non-negative -> uint-ordered
    unsigned lb = (unsigned)j;

    for (int k = 0; k < TOPK; ++k) {
        unsigned h = hb, l = lb;
        warpMinPair(h, l);
        if (lane == 0) { s_wh[wid] = h; s_wl[wid] = l; }
        __syncthreads();
        if (wid == 0) {
            unsigned h2 = s_wh[lane], l2 = s_wl[lane];
            warpMinPair(h2, l2);
            if (lane == 0) {
                Eidx[(size_t)bl * TOPK + k] = (int)l2;
                Dn  [(size_t)bl * TOPK + k] = __uint_as_float(h2);
                s_bcast = l2;
            }
        }
        __syncthreads();
        if ((unsigned)j == s_bcast) { hb = 0xFFFFFFFFu; lb = 0xFFFFFFFFu; }
        __syncthreads();
    }
}

// ---------------- kernel 4: features + WMMA GEMM + LayerNorm ----------------
__device__ inline v16h loadA(const _Float16* fbase, int row, int k0) {
    const v8h lo = *(const v8h*)(fbase + row * FSTRIDE + k0);
    const v8h hi = *(const v8h*)(fbase + row * FSTRIDE + k0 + 16);
    v16h a;
#pragma unroll
    for (int t = 0; t < 8; ++t) { a[t] = lo[t]; a[t + 8] = hi[t]; }
    return a;
}

__global__ __launch_bounds__(256) void k_edge(const float* __restrict__ coords,
                                              const _Float16* __restrict__ Wh,
                                              const float* __restrict__ Dn,
                                              const int* __restrict__ Eidx,
                                              const int* __restrict__ ridx,
                                              const int* __restrict__ chain,
                                              const float* __restrict__ W_pos,
                                              const float* __restrict__ b_pos,
                                              const float* __restrict__ ln_g,
                                              const float* __restrict__ ln_b,
                                              float* __restrict__ out,
                                              int L) {
    __shared__ __align__(16) _Float16 feat[TOPK * FSTRIDE];  // 40,704 B; reused as f32 E tile
    __shared__ float s_nbr[TOPK][16];   // 15 used
    __shared__ float s_ic[16];
    __shared__ int   s_dpos[TOPK];
    __shared__ float s_dn[TOPK];

    int bl  = blockIdx.x;
    int b   = bl / L;
    int i   = bl % L;
    int tid = threadIdx.x;
    int lane = tid & 31;
    int w    = tid >> 5;        // wave id == N-tile (0..7)
    int lgrp = lane >> 4;       // K sub-block select
    int lm   = lane & 15;       // M/N row within tile

    if (tid < TOPK) {
        int e = tid;
        int j = Eidx[(size_t)bl * TOPK + e];
        const float* cj = coords + ((size_t)(b * L + j)) * 15;
#pragma unroll
        for (int q = 0; q < 15; ++q) s_nbr[e][q] = cj[q];
        s_dn[e] = Dn[(size_t)bl * TOPK + e];
        int off  = ridx[b*L + i] - ridx[b*L + j];
        int same = (chain[b*L + i] == chain[b*L + j]) ? 1 : 0;
        int d = off + 32; d = d < 0 ? 0 : (d > 64 ? 64 : d);
        s_dpos[e] = same ? d : 65;
    } else if (tid < TOPK + 15) {
        int q = tid - TOPK;
        s_ic[q] = coords[(size_t)bl * 15 + q];
    }
    __syncthreads();

    // ---- build 48 x 416 f16 feature tile ----
    const int NT = 26 * TOPK;   // 26 groups of 16 features per edge
    for (int task = tid; task < NT; task += 256) {
        int e = task % TOPK;
        int g = task / TOPK;
        _Float16* fr = feat + e * FSTRIDE;
        if (g == 0) {
            int d = s_dpos[e];
#pragma unroll
            for (int f = 0; f < 16; ++f)
                fr[f] = (_Float16)(W_pos[f * 66 + d] + b_pos[f]);
        } else {
            float D;
            if (g == 1) {
                D = s_dn[e];
            } else {
                int aa = c_pa[g - 2], bb = c_pb[g - 2];
                float dx = s_ic[aa*3+0] - s_nbr[e][bb*3+0];
                float dy = s_ic[aa*3+1] - s_nbr[e][bb*3+1];
                float dz = s_ic[aa*3+2] - s_nbr[e][bb*3+2];
                D = sqrtf(dx*dx + dy*dy + dz*dz + 1e-6f);
            }
            int base = 16 + (g - 1) * 16;
#pragma unroll
            for (int r = 0; r < 16; ++r) {
                float mu = 2.0f + (20.0f / 15.0f) * (float)r;
                float t  = (D - mu) * 0.8f;      // /1.25
                fr[base + r] = (_Float16)__expf(-t * t);
            }
        }
    }
    __syncthreads();

    // ---- GEMM: (48x416) @ W_edge^T(416x128), f16 inputs, f32 accumulate ----
    v8f acc0 = {}, acc1 = {}, acc2 = {};
    for (int kb = 0; kb < 13; ++kb) {
        int k0 = kb * 32 + lgrp * 8;
        // B fragment: column N = w*16+lm of W_edge^T == row (w*16+lm) of W_edge
        const _Float16* wp = Wh + (size_t)(w * 16 + lm) * FEAT + k0;
        const v8h blo = *(const v8h*)(wp);
        const v8h bhi = *(const v8h*)(wp + 16);
        v16h bfrag;
#pragma unroll
        for (int t = 0; t < 8; ++t) { bfrag[t] = blo[t]; bfrag[t + 8] = bhi[t]; }

        v16h a0 = loadA(feat, 0 * 16 + lm, k0);
        v16h a1 = loadA(feat, 1 * 16 + lm, k0);
        v16h a2 = loadA(feat, 2 * 16 + lm, k0);

        acc0 = __builtin_amdgcn_wmma_f32_16x16x32_f16(false, a0, false, bfrag, (short)0, acc0, false, false);
        acc1 = __builtin_amdgcn_wmma_f32_16x16x32_f16(false, a1, false, bfrag, (short)0, acc1, false, false);
        acc2 = __builtin_amdgcn_wmma_f32_16x16x32_f16(false, a2, false, bfrag, (short)0, acc2, false, false);
    }
    __syncthreads();   // all feat reads done; reuse LDS as f32 E tile

    float* Etile = (float*)feat;   // 48*128 f32 = 24,576 B <= 40,704 B
#pragma unroll
    for (int r = 0; r < 8; ++r) {
        int mrow = lgrp * 8 + r;
        Etile[(0 * 16 + mrow) * EDIM + w * 16 + lm] = acc0[r];
        Etile[(1 * 16 + mrow) * EDIM + w * 16 + lm] = acc1[r];
        Etile[(2 * 16 + mrow) * EDIM + w * 16 + lm] = acc2[r];
    }
    __syncthreads();

    // ---- fused LayerNorm(128) + store ----
    for (int row = w; row < TOPK; row += 8) {
        float x[4];
#pragma unroll
        for (int c = 0; c < 4; ++c) x[c] = Etile[row * EDIM + lane * 4 + c];
        float s  = x[0] + x[1] + x[2] + x[3];
        float ss = x[0]*x[0] + x[1]*x[1] + x[2]*x[2] + x[3]*x[3];
#pragma unroll
        for (int off = 16; off > 0; off >>= 1) {
            s  += __shfl_xor(s,  off, 32);
            ss += __shfl_xor(ss, off, 32);
        }
        float mean = s * (1.0f / 128.0f);
        float var  = ss * (1.0f / 128.0f) - mean * mean;
        float rstd = rsqrtf(var + 1e-5f);
        size_t obase = ((size_t)bl * TOPK + row) * EDIM;
#pragma unroll
        for (int c = 0; c < 4; ++c) {
            int f = lane * 4 + c;
            out[obase + f] = (x[c] - mean) * rstd * ln_g[f] + ln_b[f];
        }
    }
}

// ---------------- launcher ----------------
extern "C" void kernel_launch(void* const* d_in, const int* in_sizes, int n_in,
                              void* d_out, int out_size, void* d_ws, size_t ws_size,
                              hipStream_t stream) {
    const float* X      = (const float*)d_in[0];
    const float* mask   = (const float*)d_in[1];
    const int*   ridx   = (const int*)d_in[2];
    const int*   chain  = (const int*)d_in[3];
    const float* W_pos  = (const float*)d_in[4];
    const float* b_pos  = (const float*)d_in[5];
    const float* W_edge = (const float*)d_in[6];
    const float* ln_g   = (const float*)d_in[7];
    const float* ln_b   = (const float*)d_in[8];

    const int L  = 1024;                  // reference geometry
    const int BL = in_sizes[1];           // B * L (mask element count)

    // workspace layout (256B aligned)
    size_t coordsBytes = (size_t)BL * 15 * sizeof(float);
    size_t off1 = (coordsBytes + 255) & ~(size_t)255;
    size_t whBytes = (size_t)EDIM * FEAT * sizeof(_Float16);
    size_t off2 = (off1 + whBytes + 255) & ~(size_t)255;

    float*     coords = (float*)d_ws;
    _Float16*  Wh     = (_Float16*)((char*)d_ws + off1);
    float*     Dnb    = (float*)((char*)d_ws + off2);

    float* Eout = (float*)d_out;
    int*   Eidx = (int*)(Eout + (size_t)BL * TOPK * EDIM);   // tuple tail: E_idx

    k_coords<<<(BL + 255) / 256, 256, 0, stream>>>(X, coords, BL);
    k_whalf <<<(EDIM * FEAT + 255) / 256, 256, 0, stream>>>(W_edge, Wh, EDIM * FEAT);
    k_topk  <<<BL, 1024, 0, stream>>>(coords, mask, Dnb, Eidx, L);
    k_edge  <<<BL, 256, 0, stream>>>(coords, Wh, Dnb, Eidx, ridx, chain,
                                     W_pos, b_pos, ln_g, ln_b, Eout, L);
}
